// BodyModel_68066641707485
// MI455X (gfx1250) — compile-verified, matrix operand order checked by reading, not run on hardware
//
#include <hip/hip_runtime.h>

// ---- problem constants -------------------------------------------------
#define B_    1024
#define V_    6890
#define J_    24
#define PF_   207          // (J-1)*9
#define KPAD  224          // PF_ padded to multiple of 32 (7 WMMA K-steps)
#define N3    20670        // V*3
#define NPAD  20672        // N3 padded to multiple of 16 (1292 tiles)
#define NT    1292
#define VT    431          // ceil(V/16) -> 6896 padded rows
#define MBLK  4            // M-tiles per wave in the big GEMM

typedef __attribute__((ext_vector_type(16))) _Float16 v16h;
typedef __attribute__((ext_vector_type(8)))  _Float16 v8h;
typedef __attribute__((ext_vector_type(8)))  float    v8f;
typedef __attribute__((ext_vector_type(4)))  int      v4i;

union HU { v16h v; v8h c[2]; };

#ifndef HAS_ASYNC_LDS
#if defined(__has_builtin)
#if __has_builtin(__builtin_amdgcn_global_load_async_to_lds_b128)
#define HAS_ASYNC_LDS 1
#endif
#endif
#endif
#ifndef HAS_ASYNC_LDS
#define HAS_ASYNC_LDS 0
#endif

// async_to_lds builtin parameter types: (v4i AS1*, v4i AS3*, imm offset, imm cpol)
typedef v4i __attribute__((address_space(1)))* gv4i_p;
typedef v4i __attribute__((address_space(3)))* lv4i_p;

__device__ __forceinline__ void wait_asynccnt0() {
#if defined(__has_builtin)
#if __has_builtin(__builtin_amdgcn_s_wait_asynccnt)
    __builtin_amdgcn_s_wait_asynccnt(0);
    return;
#else
    asm volatile("s_wait_asynccnt 0x0" ::: "memory");
    return;
#endif
#else
    asm volatile("s_wait_asynccnt 0x0" ::: "memory");
#endif
}

// ---- output layout (flat, reference return order) ----------------------
// verts   : [0,            21166080)
// J_posed : [21166080,     21239808)
// J_rest  : [21239808,     21313536)
// A       : [21313536,     21706752)
// v_shaped: [21706752,     42872832)

// ---- workspace layout (bytes) ------------------------------------------
// pdT  f16 [NPAD][KPAD]          @ 0          (9,261,056 B)
// wh   f16 [6896][32]            @ 9,261,056  (441,344 B)
// pf   f16 [B][KPAD]             @ 9,702,400  (458,752 B)
// AT   f16 [B][16][32]           @ 10,161,152 (1,048,576 B)

// ------------------------------------------------------------------------
// Convert posedirs (207 x N3 row-major f32) -> N-major padded f16 [NPAD][KPAD]
__global__ void k_cvt_pd(const float* __restrict__ pd, _Float16* __restrict__ pdT) {
    int idx = blockIdx.x * blockDim.x + threadIdx.x;      // kr*NPAD + n
    if (idx >= KPAD * NPAD) return;
    int kr = idx / NPAD;
    int n  = idx % NPAD;
    float v = (kr < PF_ && n < N3) ? pd[(long)kr * N3 + n] : 0.0f;
    pdT[(long)n * KPAD + kr] = (_Float16)v;
}

// Convert lbs_weights (V x 24 f32) -> padded f16 [6896][32]
__global__ void k_cvt_w(const float* __restrict__ w, _Float16* __restrict__ wh) {
    int idx = blockIdx.x * blockDim.x + threadIdx.x;
    if (idx >= 6896 * 32) return;
    int v = idx >> 5, j = idx & 31;
    float x = (v < V_ && j < J_) ? w[v * J_ + j] : 0.0f;
    wh[idx] = (_Float16)x;
}

// v_shaped = v_template + shapedirs . betas   (memory bound; VALU)
__global__ void k_shape(const float* __restrict__ betas, const float* __restrict__ vt,
                        const float* __restrict__ sd, float* __restrict__ vsh) {
    int idx = blockIdx.x * blockDim.x + threadIdx.x;      // b*V + v
    if (idx >= B_ * V_) return;
    int b = idx / V_, v = idx % V_;
    float be[10];
    const float* bb = betas + b * 10;
#pragma unroll
    for (int l = 0; l < 10; ++l) be[l] = bb[l];
#pragma unroll
    for (int k = 0; k < 3; ++k) {
        const float* s = sd + ((long)v * 3 + k) * 10;
        float acc = vt[v * 3 + k];
#pragma unroll
        for (int l = 0; l < 10; ++l) acc += be[l] * s[l];
        vsh[(long)b * N3 + v * 3 + k] = acc;
    }
}

// J_rest[b,j,k] = sum_v Jreg[j,v] * v_shaped[b,v,k]
__global__ void k_jrest(const float* __restrict__ jreg, const float* __restrict__ vsh,
                        float* __restrict__ jrest) {
    int b = blockIdx.x, t = threadIdx.x;
    if (t >= J_ * 3) return;
    int j = t / 3, k = t % 3;
    const float* jr = jreg + (long)j * V_;
    const float* vs = vsh + (long)b * N3 + k;
    float acc = 0.0f;
    for (int v = 0; v < V_; ++v) acc += jr[v] * vs[v * 3];
    jrest[b * (J_ * 3) + t] = acc;
}

// Per-batch: Rodrigues, pose feature (f16), kinematic chain, A matrices.
__global__ void k_pose(const float* __restrict__ pose, const int* __restrict__ parents,
                       const float* __restrict__ jrest, float* __restrict__ Aout,
                       float* __restrict__ jposed, _Float16* __restrict__ pf,
                       _Float16* __restrict__ AT) {
    __shared__ float R[J_][9];
    __shared__ float Jr[J_][3];
    __shared__ float Tch[J_][12];   // chain transforms (3x4)
    __shared__ float Al[J_][16];    // final A (4x4)
    __shared__ float Jp[J_][3];
    int b = blockIdx.x, l = threadIdx.x;

    if (l < J_) {
        float x = pose[b * 72 + l * 3 + 0] + 1e-8f;
        float y = pose[b * 72 + l * 3 + 1] + 1e-8f;
        float z = pose[b * 72 + l * 3 + 2] + 1e-8f;
        float ang = sqrtf(x * x + y * y + z * z);
        float inv = 1.0f / ang;
        float kx = x * inv, ky = y * inv, kz = z * inv;
        float s = sinf(ang), c = cosf(ang), mc = 1.0f - c;
        R[l][0] = 1.0f - mc * (ky * ky + kz * kz);
        R[l][1] = -s * kz + mc * kx * ky;
        R[l][2] =  s * ky + mc * kx * kz;
        R[l][3] =  s * kz + mc * kx * ky;
        R[l][4] = 1.0f - mc * (kx * kx + kz * kz);
        R[l][5] = -s * kx + mc * ky * kz;
        R[l][6] = -s * ky + mc * kx * kz;
        R[l][7] =  s * kx + mc * ky * kz;
        R[l][8] = 1.0f - mc * (kx * kx + ky * ky);
        Jr[l][0] = jrest[b * 72 + l * 3 + 0];
        Jr[l][1] = jrest[b * 72 + l * 3 + 1];
        Jr[l][2] = jrest[b * 72 + l * 3 + 2];
    }
    __syncthreads();

    // pose feature (rot_mats[1:] - I), K-padded to 224, f16
    if (l >= 1 && l < J_) {
#pragma unroll
        for (int e = 0; e < 9; ++e) {
            float d = R[l][e] - ((e == 0 || e == 4 || e == 8) ? 1.0f : 0.0f);
            pf[(long)b * KPAD + (l - 1) * 9 + e] = (_Float16)d;
        }
    }
    if (l == 0)
        for (int e = PF_; e < KPAD; ++e) pf[(long)b * KPAD + e] = (_Float16)0.0f;

    // sequential kinematic chain (lane 0)
    if (l == 0) {
        for (int m = 0; m < 3; ++m) {
            for (int q = 0; q < 3; ++q) Tch[0][m * 4 + q] = R[0][m * 3 + q];
            Tch[0][m * 4 + 3] = Jr[0][m];
        }
        for (int i = 1; i < J_; ++i) {
            int p = parents[i];
            float rel0 = Jr[i][0] - Jr[p][0];
            float rel1 = Jr[i][1] - Jr[p][1];
            float rel2 = Jr[i][2] - Jr[p][2];
            for (int m = 0; m < 3; ++m) {
                for (int n = 0; n < 3; ++n)
                    Tch[i][m * 4 + n] = Tch[p][m * 4 + 0] * R[i][0 * 3 + n]
                                      + Tch[p][m * 4 + 1] * R[i][1 * 3 + n]
                                      + Tch[p][m * 4 + 2] * R[i][2 * 3 + n];
                Tch[i][m * 4 + 3] = Tch[p][m * 4 + 0] * rel0
                                  + Tch[p][m * 4 + 1] * rel1
                                  + Tch[p][m * 4 + 2] * rel2
                                  + Tch[p][m * 4 + 3];
            }
        }
        for (int j = 0; j < J_; ++j) {
            for (int m = 0; m < 3; ++m) {
                Jp[j][m] = Tch[j][m * 4 + 3];
                float rj = Tch[j][m * 4 + 0] * Jr[j][0]
                         + Tch[j][m * 4 + 1] * Jr[j][1]
                         + Tch[j][m * 4 + 2] * Jr[j][2];
                Al[j][m * 4 + 0] = Tch[j][m * 4 + 0];
                Al[j][m * 4 + 1] = Tch[j][m * 4 + 1];
                Al[j][m * 4 + 2] = Tch[j][m * 4 + 2];
                Al[j][m * 4 + 3] = Tch[j][m * 4 + 3] - rj;
            }
            Al[j][12] = 0.0f; Al[j][13] = 0.0f; Al[j][14] = 0.0f; Al[j][15] = 1.0f;
        }
    }
    __syncthreads();

    if (l < J_) {
#pragma unroll
        for (int e = 0; e < 16; ++e) Aout[(long)b * 384 + l * 16 + e] = Al[l][e];
        jposed[b * 72 + l * 3 + 0] = Jp[l][0];
        jposed[b * 72 + l * 3 + 1] = Jp[l][1];
        jposed[b * 72 + l * 3 + 2] = Jp[l][2];
    }
    // transposed f16 A for WMMA B-operand: AT[b][n=0..15][k=0..31]
    if (l < 16) {
        _Float16* row = AT + (long)b * 512 + l * 32;
        for (int k = 0; k < J_; ++k) row[k] = (_Float16)Al[k][l];
        for (int k = J_; k < 32; ++k) row[k] = (_Float16)0.0f;
    }
}

// v_posed = v_shaped + pose_feature @ posedirs  (M=1024, N=20670, K=224)
// One wave computes MBLK=4 vertically-stacked 16x16 tiles, reusing the B
// fragment 4x. The 8 waves of a block share one ntile (same B stream).
// 7 K-steps x 4 M-tiles = 28 WMMAs per wave. Writes v_posed into verts slot.
__global__ void k_gemm_pose(const _Float16* __restrict__ pf, const _Float16* __restrict__ pdT,
                            const float* __restrict__ vsh, float* __restrict__ vposed) {
    int wid  = threadIdx.x >> 5;
    int lane = threadIdx.x & 31;
    int gw = blockIdx.x * 8 + wid;
    int ntile  = gw >> 4;                 // 0..1291  (16 waves per ntile)
    int mgroup = gw & 15;                 // rows [mgroup*64, mgroup*64+64)
    int half_sel = (lane < 16) ? 0 : 8;   // K-chunk select AND D-row offset
    int rloc = lane & 15;

    const _Float16* brow  = pdT + (long)(ntile * 16 + rloc) * KPAD;
    const _Float16* arow0 = pf  + (long)(mgroup * 64 + rloc) * KPAD;

    v8f zero = {0.f, 0.f, 0.f, 0.f, 0.f, 0.f, 0.f, 0.f};
    v8f acc[MBLK];
#pragma unroll
    for (int m = 0; m < MBLK; ++m) acc[m] = zero;

#pragma unroll
    for (int s = 0; s < 7; ++s) {
        int k0 = s * 32 + half_sel;
        HU bm;
        bm.c[0] = *(const v8h*)(brow + k0);
        bm.c[1] = *(const v8h*)(brow + k0 + 16);
#pragma unroll
        for (int m = 0; m < MBLK; ++m) {
            HU a;
            const _Float16* arow = arow0 + (long)m * 16 * KPAD;
            a.c[0] = *(const v8h*)(arow + k0);
            a.c[1] = *(const v8h*)(arow + k0 + 16);
            acc[m] = __builtin_amdgcn_wmma_f32_16x16x32_f16(
                false, a.v, false, bm.v, (short)0, acc[m], false, false);
        }
    }
    int col = ntile * 16 + rloc;
    if (col < N3) {
#pragma unroll
        for (int m = 0; m < MBLK; ++m) {
#pragma unroll
            for (int r = 0; r < 8; ++r) {
                long row = (long)(mgroup * 64 + m * 16 + r + half_sel);
                long off = row * N3 + col;
                vposed[off] = acc[m][r] + vsh[off];
            }
        }
    }
}

// Fused LBS: T = lbsw_tile(16x32) @ A_T(32x16) via one WMMA, then apply T
// to v_posed in place (verts buffer holds v_posed on entry, verts on exit).
// The per-batch A_T tile (1 KB) is staged to LDS with the gfx1250 async
// DMA path when available.
__global__ void k_lbs(const _Float16* __restrict__ wh, const _Float16* __restrict__ AT,
                      float* __restrict__ verts) {
    __shared__ float tl[8][16][16];
#if HAS_ASYNC_LDS
    __shared__ _Float16 astage[8][16][32];
#endif
    int wid  = threadIdx.x >> 5;
    int lane = threadIdx.x & 31;
    int gw = blockIdx.x * 8 + wid;
    int b = gw / VT;
    int vtile = gw % VT;
    int half_sel = (lane < 16) ? 0 : 8;
    int rloc = lane & 15;

    const _Float16* arow = wh + (long)(vtile * 16 + rloc) * 32;
    HU a, bm;
    a.c[0] = *(const v8h*)(arow + half_sel);
    a.c[1] = *(const v8h*)(arow + half_sel + 16);

#if HAS_ASYNC_LDS
    {
        const _Float16* ATb = AT + (long)b * 512;
        _Float16* dst0 = &astage[wid][0][0];
        // 1 KB tile = 64 x 16B chunks; 32 lanes x 2 async b128 transfers.
        __builtin_amdgcn_global_load_async_to_lds_b128(
            (gv4i_p)(ATb + lane * 8),
            (lv4i_p)(dst0 + lane * 8), 0, 0);
        __builtin_amdgcn_global_load_async_to_lds_b128(
            (gv4i_p)(ATb + 256 + lane * 8),
            (lv4i_p)(dst0 + 256 + lane * 8), 0, 0);
        wait_asynccnt0();
        bm.c[0] = *(const v8h*)(&astage[wid][rloc][half_sel]);
        bm.c[1] = *(const v8h*)(&astage[wid][rloc][half_sel + 16]);
    }
#else
    {
        const _Float16* brow = AT + (long)b * 512 + rloc * 32;
        bm.c[0] = *(const v8h*)(brow + half_sel);
        bm.c[1] = *(const v8h*)(brow + half_sel + 16);
    }
#endif

    v8f acc = {0.f, 0.f, 0.f, 0.f, 0.f, 0.f, 0.f, 0.f};
    acc = __builtin_amdgcn_wmma_f32_16x16x32_f16(
        false, a.v, false, bm.v, (short)0, acc, false, false);

#pragma unroll
    for (int r = 0; r < 8; ++r) tl[wid][r + half_sel][rloc] = acc[r];
    __syncthreads();

    if (lane < 16) {
        int v = vtile * 16 + lane;
        if (v < V_) {
            long base = (long)b * N3 + v * 3;
            float x = verts[base + 0];
            float y = verts[base + 1];
            float z = verts[base + 2];
            const float* Tr = tl[wid][lane];
#pragma unroll
            for (int m = 0; m < 3; ++m)
                verts[base + m] = Tr[m * 4 + 0] * x + Tr[m * 4 + 1] * y
                                + Tr[m * 4 + 2] * z + Tr[m * 4 + 3];
        }
    }
}

extern "C" void kernel_launch(void* const* d_in, const int* in_sizes, int n_in,
                              void* d_out, int out_size, void* d_ws, size_t ws_size,
                              hipStream_t stream) {
    const float* betas   = (const float*)d_in[0];
    const float* pose    = (const float*)d_in[1];
    const float* vtmpl   = (const float*)d_in[2];
    const float* sdirs   = (const float*)d_in[3];
    const float* pdirs   = (const float*)d_in[4];
    const float* jreg    = (const float*)d_in[5];
    const float* lbsw    = (const float*)d_in[6];
    const int*   parents = (const int*)d_in[7];

    float* out    = (float*)d_out;
    float* verts  = out;                       // 21,166,080
    float* jposed = out + 21166080;            //     73,728
    float* jrest  = jposed + 73728;            //     73,728
    float* Aout   = jrest + 73728;             //    393,216
    float* vsh    = Aout + 393216;             // 21,166,080

    char* ws = (char*)d_ws;
    _Float16* pdT = (_Float16*)(ws + 0);
    _Float16* wh  = (_Float16*)(ws + 9261056);
    _Float16* pf  = (_Float16*)(ws + 9702400);
    _Float16* AT  = (_Float16*)(ws + 10161152);

    k_cvt_pd   <<<(KPAD * NPAD + 255) / 256, 256, 0, stream>>>(pdirs, pdT);
    k_cvt_w    <<<(6896 * 32 + 255) / 256,   256, 0, stream>>>(lbsw, wh);
    k_shape    <<<(B_ * V_ + 255) / 256,     256, 0, stream>>>(betas, vtmpl, sdirs, vsh);
    k_jrest    <<<B_, 128, 0, stream>>>(jreg, vsh, jrest);
    k_pose     <<<B_,  32, 0, stream>>>(pose, parents, jrest, Aout, jposed, pf, AT);
    k_gemm_pose<<<(NT * 16) / 8, 256, 0, stream>>>(pf, pdT, vsh, verts);
    k_lbs      <<<(B_ * VT) / 8, 256, 0, stream>>>(wh, AT, verts);
}
